// BrickTube_76579266887845
// MI455X (gfx1250) — compile-verified, matrix operand order checked by reading, not run on hardware
//
#include <hip/hip_runtime.h>
#include <hip/hip_bf16.h>

// ---------------------------------------------------------------------------
// BrickTube = 80 two-qubit gates acting linearly on a 1024-dim state.
// => out[32768,1024] = x[32768,1024] @ M[1024,1024].
// Phase 1: build M from the Kronecker gate structure (cheap, ~0.7 GFLOP).
// Phase 2: GEMM via split-precision bf16 WMMA (hi/lo decomposition, 3 terms,
//          f32 accumulate), double-buffered LDS, async global->LDS staging
//          for the B operand (GLOBAL_LOAD_ASYNC_TO_LDS_B128, ASYNCcnt).
// ---------------------------------------------------------------------------

typedef __attribute__((ext_vector_type(16))) __bf16          v16bf;
typedef __attribute__((ext_vector_type(8)))  float           v8f;
typedef __attribute__((ext_vector_type(8)))  unsigned short  us8;

union Frag16 {
    v16bf v;
    us8   h[2];
};

__device__ __forceinline__ unsigned short f32_to_bf16_rne(float f) {
    unsigned int u = __float_as_uint(f);
    unsigned int r = u + 0x7FFFu + ((u >> 16) & 1u);   // round-to-nearest-even
    return (unsigned short)(r >> 16);
}
__device__ __forceinline__ float bf16_to_f32(unsigned short h) {
    return __uint_as_float(((unsigned int)h) << 16);
}

// ---------------------------------------------------------------------------
// Phase 1: M^T (hi/lo bf16) from the gate network applied to identity rows.
// Qubit axis i <-> bit (9-i) of the flat 1024 index (MSB-first reshape).
// ---------------------------------------------------------------------------
__global__ __launch_bounds__(256) void build_mt_kernel(
    const float* __restrict__ cores,        // [80][2][2][2][2]
    unsigned short* __restrict__ mt_hi,     // [1024][1024]  Mt[c][r] = M[r][c]
    unsigned short* __restrict__ mt_lo)
{
    __shared__ float s[8][1024];            // 8 rows of M per block (32 KB)
    const int tid = threadIdx.x;
    const int r0  = blockIdx.x * 8;

    for (int idx = tid; idx < 8 * 1024; idx += 256) {
        int row = idx >> 10, c = idx & 1023;
        s[row][c] = (c == r0 + row) ? 1.0f : 0.0f;
    }
    __syncthreads();

    for (int layer = 0; layer < 8; ++layer) {
        for (int sweep = 0; sweep < 2; ++sweep) {
            for (int g5 = 0; g5 < 5; ++g5) {
                int qi, qj, ci;
                if (sweep == 0) { qi = 2 * g5;     qj = 2 * g5 + 1;        ci = layer * 10 + g5; }
                else            { qi = 2 * g5 + 1; qj = (2 * g5 + 2) % 10; ci = layer * 10 + 5 + g5; }
                const int b0 = 9 - qi;              // bit for qubit qi
                const int b1 = 9 - qj;              // bit for qubit qj
                const unsigned m0 = 1u << b0, m1 = 1u << b1;
                const float* cp = cores + ci * 16;  // uniform -> scalar loads
                float c[16];
                #pragma unroll
                for (int t = 0; t < 16; ++t) c[t] = cp[t];

                for (int task = tid; task < 2048; task += 256) {
                    const int row = task >> 8;
                    unsigned gg = (unsigned)(task & 255);
                    unsigned base = 0;
                    #pragma unroll
                    for (int b = 0; b < 10; ++b) {
                        if (b == b0 || b == b1) continue;
                        base |= (gg & 1u) << b;
                        gg >>= 1;
                    }
                    const float v00 = s[row][base];
                    const float v01 = s[row][base | m1];
                    const float v10 = s[row][base | m0];
                    const float v11 = s[row][base | m0 | m1];
                    // new[n0][n1] = sum_{x0,x1} v[x0x1] * core[x0][x1][n0][n1]
                    s[row][base]           = v00*c[0] + v01*c[4] + v10*c[8]  + v11*c[12];
                    s[row][base | m1]      = v00*c[1] + v01*c[5] + v10*c[9]  + v11*c[13];
                    s[row][base | m0]      = v00*c[2] + v01*c[6] + v10*c[10] + v11*c[14];
                    s[row][base | m0 | m1] = v00*c[3] + v01*c[7] + v10*c[11] + v11*c[15];
                }
                __syncthreads();
            }
        }
    }

    // Write transposed hi/lo split: Mt[c][r]
    for (int idx = tid; idx < 8 * 1024; idx += 256) {
        int row = idx >> 10, c = idx & 1023;
        float f = s[row][c];
        unsigned short hi = f32_to_bf16_rne(f);
        unsigned short lo = f32_to_bf16_rne(f - bf16_to_f32(hi));
        size_t o = (size_t)c * 1024 + (size_t)(r0 + row);
        mt_hi[o] = hi;
        mt_lo[o] = lo;
    }
}

// ---------------------------------------------------------------------------
// Phase 2: C = X * M with split-bf16 WMMA.
// Block: 512 threads (16 waves), tile 128(M) x 256(N), K-step 32, LDS double-
// buffered. B operand staged with GLOBAL_LOAD_ASYNC_TO_LDS_B128 (raw bf16
// copy, ASYNCcnt); A operand split f32->bf16 hi/lo by VALU while staging.
// ---------------------------------------------------------------------------
#define TM 128
#define TN 256
#define TK 32

__device__ __forceinline__ void stage_X_split(
    unsigned short (*Xh)[TK], unsigned short (*Xl)[TK],
    const float* __restrict__ X, int row0, int k0, int tid)
{
    const int r  = tid >> 2;             // 0..127
    const int kb = (tid & 3) * 8;        // 0,8,16,24
    const float* src = X + (size_t)(row0 + r) * 1024 + k0 + kb;
    float4 f0 = *(const float4*)(src);
    float4 f1 = *(const float4*)(src + 4);
    float vals[8] = {f0.x, f0.y, f0.z, f0.w, f1.x, f1.y, f1.z, f1.w};
    #pragma unroll
    for (int e = 0; e < 8; ++e) {
        unsigned short hi = f32_to_bf16_rne(vals[e]);
        Xh[r][kb + e] = hi;
        Xl[r][kb + e] = f32_to_bf16_rne(vals[e] - bf16_to_f32(hi));
    }
    if (k0 + TK < 1024)                  // hint next X tile toward caches
        __builtin_prefetch(src + TK, 0, 1);
}

// Async DMA of one 256x32 bf16 tile of Mt into LDS (hi and lo planes).
// GVS mode: mem = SADDR(64) + VADDR(32) + IOFFSET; IOFFSET advances both the
// memory address and the LDS address, so offset:16 is the second 16B chunk.
__device__ __forceinline__ void stage_B_async(
    unsigned short (*Bh)[TK], unsigned short (*Bl)[TK],
    const unsigned short* __restrict__ mt_hi,
    const unsigned short* __restrict__ mt_lo,
    int col0, int k0, int tid)
{
    const int n  = tid >> 1;             // 0..255
    const int kb = (tid & 1) * 16;       // 0,16
    const unsigned voff = (((unsigned)(col0 + n) << 10) + (unsigned)(k0 + kb)) * 2u;
    const unsigned ldsh = (unsigned)(uintptr_t)&Bh[n][kb];
    const unsigned ldsl = (unsigned)(uintptr_t)&Bl[n][kb];
    asm volatile("global_load_async_to_lds_b128 %0, %1, %2\n\t"
                 "global_load_async_to_lds_b128 %0, %1, %2 offset:16"
                 :: "v"(ldsh), "v"(voff), "s"(mt_hi) : "memory");
    asm volatile("global_load_async_to_lds_b128 %0, %1, %2\n\t"
                 "global_load_async_to_lds_b128 %0, %1, %2 offset:16"
                 :: "v"(ldsl), "v"(voff), "s"(mt_lo) : "memory");
}

__global__ __launch_bounds__(512) void bricktube_gemm_kernel(
    const float* __restrict__ X,                 // [32768][1024] f32
    const unsigned short* __restrict__ mt_hi,    // [1024][1024] bf16, Mt[n][k]
    const unsigned short* __restrict__ mt_lo,
    float* __restrict__ out)                     // [32768][1024] f32
{
    __shared__ unsigned short Xs_hi[2][TM][TK];  // 16 KB
    __shared__ unsigned short Xs_lo[2][TM][TK];  // 16 KB
    __shared__ unsigned short Bs_hi[2][TN][TK];  // 32 KB
    __shared__ unsigned short Bs_lo[2][TN][TK];  // 32 KB  (96 KB of 320 KB/WGP)

    const int tid  = threadIdx.x;
    const int wave = tid >> 5;
    const int lane = tid & 31;
    const int hf   = lane >> 4;                  // half of wave (0/1)
    const int l16  = lane & 15;
    const int rowgrp = wave & 7;                 // 0..7  -> 16-row strip
    const int colgrp = wave >> 3;                // 0..1  -> 128-col half
    // Column-blocks on blockIdx.x (fast): the 4 column passes over one X
    // row-strip are dispatched adjacently -> X is read from HBM ~once (L2 hit).
    const int col0 = blockIdx.x * TN;
    const int row0 = blockIdx.y * TM;

    v8f acc[8];
    #pragma unroll
    for (int t = 0; t < 8; ++t)
        #pragma unroll
        for (int i = 0; i < 8; ++i) acc[t][i] = 0.0f;

    // Prologue: fill buffer 0.
    stage_B_async(Bs_hi[0], Bs_lo[0], mt_hi, mt_lo, col0, 0, tid);
    stage_X_split(Xs_hi[0], Xs_lo[0], X, row0, 0, tid);
    asm volatile("s_wait_asynccnt 0x0" ::: "memory");
    __syncthreads();

    int buf = 0;
    for (int k0 = 0; k0 < 1024; k0 += TK, buf ^= 1) {
        // Prefetch next K-step into the back buffer while computing.
        if (k0 + TK < 1024) {
            stage_B_async(Bs_hi[buf ^ 1], Bs_lo[buf ^ 1], mt_hi, mt_lo,
                          col0, k0 + TK, tid);
            stage_X_split(Xs_hi[buf ^ 1], Xs_lo[buf ^ 1], X, row0, k0 + TK, tid);
        }

        // ---- A fragments (ISA 16-bit A layout: lane->M row, VGPR0-3:
        //      K=8*hf+0..7, VGPR4-7: K=16+8*hf+0..7) ----
        const int am = 16 * rowgrp + l16;
        Frag16 ahi, alo;
        ahi.h[0] = *(const us8*)&Xs_hi[buf][am][8 * hf];
        ahi.h[1] = *(const us8*)&Xs_hi[buf][am][16 + 8 * hf];
        alo.h[0] = *(const us8*)&Xs_lo[buf][am][8 * hf];
        alo.h[1] = *(const us8*)&Xs_lo[buf][am][16 + 8 * hf];

        #pragma unroll
        for (int t = 0; t < 8; ++t) {
            // B fragment: mirrored layout (lane -> N column, same K striping
            // as A). Mt stored N-major so this is two b128 LDS reads.
            const int bn = 128 * colgrp + 16 * t + l16;
            Frag16 bhi, blo;
            bhi.h[0] = *(const us8*)&Bs_hi[buf][bn][8 * hf];
            bhi.h[1] = *(const us8*)&Bs_hi[buf][bn][16 + 8 * hf];
            blo.h[0] = *(const us8*)&Bs_lo[buf][bn][8 * hf];
            blo.h[1] = *(const us8*)&Bs_lo[buf][bn][16 + 8 * hf];

            acc[t] = __builtin_amdgcn_wmma_f32_16x16x32_bf16(
                false, ahi.v, false, bhi.v, (short)0, acc[t], false, false);
            acc[t] = __builtin_amdgcn_wmma_f32_16x16x32_bf16(
                false, ahi.v, false, blo.v, (short)0, acc[t], false, false);
            acc[t] = __builtin_amdgcn_wmma_f32_16x16x32_bf16(
                false, alo.v, false, bhi.v, (short)0, acc[t], false, false);
        }

        // Own async copies for the back buffer must land before the barrier.
        asm volatile("s_wait_asynccnt 0x0" ::: "memory");
        __syncthreads();
    }

    // ---- write back: C/D layout: VGPR v -> row v (lanes 0-15) / v+8 (16-31),
    //      N = lane%16 ----
    #pragma unroll
    for (int t = 0; t < 8; ++t) {
        const int n = col0 + 128 * colgrp + 16 * t + l16;
        #pragma unroll
        for (int v = 0; v < 8; ++v) {
            const int r = row0 + 16 * rowgrp + v + 8 * hf;
            out[(size_t)r * 1024 + n] = acc[t][v];
        }
    }
}

extern "C" void kernel_launch(void* const* d_in, const int* in_sizes, int n_in,
                              void* d_out, int out_size, void* d_ws, size_t ws_size,
                              hipStream_t stream) {
    const float* x     = (const float*)d_in[0];   // [32768*1024]
    const float* cores = (const float*)d_in[1];   // [80*16]
    float* out = (float*)d_out;                   // [32768*1024]

    unsigned short* mt_hi = (unsigned short*)d_ws;          // 2 MB
    unsigned short* mt_lo = mt_hi + (size_t)1024 * 1024;    // 2 MB

    build_mt_kernel<<<128, 256, 0, stream>>>(cores, mt_hi, mt_lo);

    dim3 grid(1024 / TN, 32768 / TM);             // 4 x 256 (cols fast)
    bricktube_gemm_kernel<<<grid, 512, 0, stream>>>(x, mt_hi, mt_lo, out);
}
